// FullQKAttention_18133351923837
// MI455X (gfx1250) — compile-verified, matrix operand order checked by reading, not run on hardware
//
#include <hip/hip_runtime.h>
#include <hip/hip_bf16.h>

typedef __attribute__((ext_vector_type(16))) _Float16 v16h;
typedef __attribute__((ext_vector_type(8)))  _Float16 v8h;
typedef __attribute__((ext_vector_type(2)))  _Float16 v2h;
typedef __attribute__((ext_vector_type(8)))  float    v8f;

#define Bq 2
#define Sq 4096
#define Dq 64
#define SCALE 0.125f               // 64^-0.5
#define TOKEN_SELF (-50000.0f)
#define MASKEDV (-3.4028234663852886e+38f)
#define JP 4                        // j-partitions for pass2
#define JCH (Sq / JP)

// ---- helpers ---------------------------------------------------------------

// Load A (16x32 f16) chunk for this lane from an f32 row pointer.
// A layout: lane l holds row M=l%16; K = {8h..8h+7} and {16+8h..16+8h+7} (h = l>=16).
__device__ inline v16h load_a_f32(const float* __restrict__ qrow, int h, int kbase) {
    v16h a;
#pragma unroll
    for (int t = 0; t < 8; ++t) {
        a[t]     = (_Float16)qrow[kbase + 8 * h + t];
        a[8 + t] = (_Float16)qrow[kbase + 16 + 8 * h + t];
    }
    return a;
}

// Load B (32x16 f16) chunk for this lane from an f16 pointer (contiguous in K).
// B layout mirrors A: lane l holds col N=l%16; K runs {8h..8h+7}, {16+8h..+7}.
__device__ inline v16h load_b_f16(const _Float16* __restrict__ p, int h) {
    const v8h lo = *(const v8h*)(p + 8 * h);
    const v8h hi = *(const v8h*)(p + 16 + 8 * h);
    v16h b;
#pragma unroll
    for (int t = 0; t < 8; ++t) { b[t] = lo[t]; b[8 + t] = hi[t]; }
    return b;
}

// ---- kernel 0: zero the `out` region (harness poisons d_out) ---------------
__global__ void zero_out_kernel(float* __restrict__ outp, int n) {
    int idx = blockIdx.x * blockDim.x + threadIdx.x;
    if (idx < n) outp[idx] = 0.0f;
}

// ---- kernel 1: k-normalization + f16 staging -------------------------------
// One wave per row: kh[row][e] = f16(qk[row][e]/max(||qk row||,eps)); vt = v^T in f16.
__global__ __launch_bounds__(32) void prep_kernel(const float* __restrict__ qk,
                                                  const float* __restrict__ v,
                                                  _Float16* __restrict__ kh,
                                                  _Float16* __restrict__ vt) {
    const int row  = blockIdx.x;          // 0 .. B*S-1
    const int bz   = row >> 12;           // / 4096
    const int ii   = row & (Sq - 1);
    const int lane = threadIdx.x;

    const float2 q2 = ((const float2*)(qk + (size_t)row * Dq))[lane];
    float ss = q2.x * q2.x + q2.y * q2.y;
#pragma unroll
    for (int off = 16; off >= 1; off >>= 1) ss += __shfl_xor(ss, off, 32);
    const float invn = 1.0f / fmaxf(sqrtf(ss), 1e-12f);

    v2h o;
    o[0] = (_Float16)(q2.x * invn);
    o[1] = (_Float16)(q2.y * invn);
    ((v2h*)(kh + (size_t)row * Dq))[lane] = o;

    const float2 v2 = ((const float2*)(v + (size_t)row * Dq))[lane];
    vt[((size_t)(bz * Dq + 2 * lane + 0)) * Sq + ii] = (_Float16)v2.x;
    vt[((size_t)(bz * Dq + 2 * lane + 1)) * Sq + ii] = (_Float16)v2.y;
}

// ---- kernel 2: row max / row sum-exp via online softmax --------------------
// 4 waves/block, each wave owns one 16-row block and streams all columns.
__global__ __launch_bounds__(128) void stats_kernel(const float* __restrict__ qk,
                                                    const int* __restrict__ maskp,
                                                    const _Float16* __restrict__ kh,
                                                    float* __restrict__ rowM,
                                                    float* __restrict__ rowS) {
    const int w    = threadIdx.x >> 5;
    const int lane = threadIdx.x & 31;
    const int h    = lane >> 4;
    const int m    = lane & 15;
    const int rb   = blockIdx.x * 4 + w;          // 0 .. 511
    const int bz   = rb >> 8;                     // / 256 row-blocks per batch
    const int i0   = (rb & 255) << 4;

    // Preload A (q tile, unnormalized): two K-chunks of 32.
    const float* qrow = qk + ((size_t)bz * Sq + i0 + m) * Dq;
    v16h a0 = load_a_f32(qrow, h, 0);
    v16h a1 = load_a_f32(qrow, h, 32);

    bool mi[8];
#pragma unroll
    for (int r = 0; r < 8; ++r) mi[r] = maskp[(size_t)bz * Sq + i0 + 8 * h + r] != 0;

    float mrun[8], srun[8];
#pragma unroll
    for (int r = 0; r < 8; ++r) { mrun[r] = MASKEDV; srun[r] = 0.0f; }

    for (int j0 = 0; j0 < Sq; j0 += 16) {
        float x[8];
        if (j0 <= i0 + 15) {                      // wave-uniform: tile touches j<=i
            const bool mj = maskp[(size_t)bz * Sq + j0 + m] != 0;
            const _Float16* kb = kh + ((size_t)bz * Sq + j0 + m) * Dq;
            v16h b0 = load_b_f16(kb, h);
            v16h b1 = load_b_f16(kb + 32, h);
            v8f acc = {};
            acc = __builtin_amdgcn_wmma_f32_16x16x32_f16(false, a0, false, b0,
                                                         (short)0, acc, false, false);
            acc = __builtin_amdgcn_wmma_f32_16x16x32_f16(false, a1, false, b1,
                                                         (short)0, acc, false, false);
#pragma unroll
            for (int r = 0; r < 8; ++r) {
                const int i = i0 + r + 8 * h;
                const int j = j0 + m;
                float xx = acc[r] * SCALE;
                if (j == i) xx = TOKEN_SELF;
                if (!(mi[r] && mj)) xx = MASKEDV;
                if (j > i) xx = MASKEDV;
                x[r] = xx;
            }
        } else {                                  // fully causal-masked tile
#pragma unroll
            for (int r = 0; r < 8; ++r) x[r] = MASKEDV;
        }
#pragma unroll
        for (int r = 0; r < 8; ++r) {             // online max/sum
            const float mo = mrun[r], xm = x[r];
            const float mn = fmaxf(mo, xm);
            srun[r] = srun[r] * __expf(mo - mn) + __expf(xm - mn);
            mrun[r] = mn;
        }
    }

    // Reduce the 16 column-partials per row (xor within each 16-lane half).
#pragma unroll
    for (int r = 0; r < 8; ++r) {
#pragma unroll
        for (int off = 8; off >= 1; off >>= 1) {
            const float mo = __shfl_xor(mrun[r], off, 32);
            const float so = __shfl_xor(srun[r], off, 32);
            const float mn = fmaxf(mrun[r], mo);
            srun[r] = srun[r] * __expf(mrun[r] - mn) + so * __expf(mo - mn);
            mrun[r] = mn;
        }
        if (m == r) {
            const int i = i0 + r + 8 * h;
            rowM[(size_t)bz * Sq + i] = mrun[r];
            rowS[(size_t)bz * Sq + i] = srun[r];
        }
    }
}

// ---- kernel 3: write attn (async LDS->global), accumulate out = attn @ v ---
// Each wave: one 16-row block x one 1024-column slice (2048 waves total).
__global__ __launch_bounds__(128) void attn_out_kernel(const float* __restrict__ qk,
                                                       const int* __restrict__ maskp,
                                                       const _Float16* __restrict__ kh,
                                                       const _Float16* __restrict__ vt,
                                                       const float* __restrict__ rowM,
                                                       const float* __restrict__ rowS,
                                                       float* __restrict__ outp,
                                                       float* __restrict__ attnp) {
    __shared__ _Float16 ldsH[4 * 16 * 32];        // per-wave 16x32 prob tile (f16, GEMM2 A)
    __shared__ float    ldsF[2 * 4 * 16 * 32];    // double-buffered f32 tile for async store

    const int w    = threadIdx.x >> 5;
    const int lane = threadIdx.x & 31;
    const int h    = lane >> 4;
    const int m    = lane & 15;
    const int wid  = blockIdx.x * 4 + w;          // 0 .. 2047
    const int rb   = wid >> 2;                    // row block 0..511
    const int jp   = wid & (JP - 1);
    const int bz   = rb >> 8;
    const int i0   = (rb & 255) << 4;
    const int jlo  = jp * JCH;

    // async-store lane mapping: 4 insts x (4 rows x 8 segs of 4 floats) per tile
    const int rsub = lane >> 3;                   // 0..3
    const int cseg = (lane & 7) * 4;              // 0,4,...,28

    const float* qrow = qk + ((size_t)bz * Sq + i0 + m) * Dq;
    v16h a0 = load_a_f32(qrow, h, 0);
    v16h a1 = load_a_f32(qrow, h, 32);

    bool  mi[8];
    float rm[8], rsi[8];
#pragma unroll
    for (int r = 0; r < 8; ++r) {
        const int i = i0 + r + 8 * h;
        mi[r]  = maskp[(size_t)bz * Sq + i0 + 8 * h + r] != 0;
        rm[r]  = rowM[(size_t)bz * Sq + i];
        rsi[r] = 1.0f / rowS[(size_t)bz * Sq + i];
    }

    v8f acc_out[4];
#pragma unroll
    for (int et = 0; et < 4; ++et) acc_out[et] = (v8f){};

    _Float16* myldH = ldsH + w * 512;

    for (int j0 = jlo; j0 < jlo + JCH; j0 += 32) {
        const int db = (j0 >> 5) & 1;
        float* myldF = ldsF + (db * 4 + w) * 512;

        // Ensure the f32 buffer we are about to overwrite has been drained
        // (at most the previous iteration's 4 stores may remain in flight).
        asm volatile("s_wait_asynccnt 0x4" ::: "memory");

#pragma unroll
        for (int jj = 0; jj < 32; jj += 16) {     // two 16-col sub-tiles
            float x[8];
            if (j0 + jj <= i0 + 15) {
                const bool mj = maskp[(size_t)bz * Sq + j0 + jj + m] != 0;
                const _Float16* kb = kh + ((size_t)bz * Sq + j0 + jj + m) * Dq;
                v16h b0 = load_b_f16(kb, h);
                v16h b1 = load_b_f16(kb + 32, h);
                v8f acc = {};
                acc = __builtin_amdgcn_wmma_f32_16x16x32_f16(false, a0, false, b0,
                                                             (short)0, acc, false, false);
                acc = __builtin_amdgcn_wmma_f32_16x16x32_f16(false, a1, false, b1,
                                                             (short)0, acc, false, false);
#pragma unroll
                for (int r = 0; r < 8; ++r) {
                    const int i = i0 + r + 8 * h;
                    const int j = j0 + jj + m;
                    float xx = acc[r] * SCALE;
                    if (j == i) xx = TOKEN_SELF;
                    if (!(mi[r] && mj)) xx = MASKEDV;
                    if (j > i) xx = MASKEDV;
                    x[r] = xx;
                }
            } else {
#pragma unroll
                for (int r = 0; r < 8; ++r) x[r] = MASKEDV;
            }
#pragma unroll
            for (int r = 0; r < 8; ++r) {
                const float p = __expf(x[r] - rm[r]) * rsi[r];
                myldF[(r + 8 * h) * 32 + jj + m] = p;           // f32 for async store
                myldH[(r + 8 * h) * 32 + jj + m] = (_Float16)p; // f16 for GEMM2 A
            }
        }
        __syncthreads();

        // Async-store the f32 prob tile LDS -> global attn (b128 per lane).
        asm volatile("s_wait_dscnt 0x0" ::: "memory");
#pragma unroll
        for (int q = 0; q < 4; ++q) {
            const int row = 4 * q + rsub;         // tile row 0..15
            unsigned ldsoff = (unsigned)(uintptr_t)&myldF[row * 32 + cseg];
            const float* gp = attnp + ((size_t)bz * Sq + i0 + row) * Sq + j0 + cseg;
            unsigned long long ga = (unsigned long long)(uintptr_t)gp;
            asm volatile("global_store_async_from_lds_b128 %0, %1, off"
                         :: "v"(ga), "v"(ldsoff) : "memory");
        }

        // Re-read prob tile in A layout (lane=row), K = 32 columns of this j0 step.
        v16h ap;
        {
            const v8h lo = *(const v8h*)(myldH + m * 32 + 8 * h);
            const v8h hi = *(const v8h*)(myldH + m * 32 + 16 + 8 * h);
#pragma unroll
            for (int t = 0; t < 8; ++t) { ap[t] = lo[t]; ap[8 + t] = hi[t]; }
        }

#pragma unroll
        for (int et = 0; et < 4; ++et) {          // out N-tiles: dims 16*et..+15
            const _Float16* vb = vt + ((size_t)(bz * Dq + 16 * et + m)) * Sq + j0;
            v16h bb = load_b_f16(vb, h);
            acc_out[et] = __builtin_amdgcn_wmma_f32_16x16x32_f16(false, ap, false, bb,
                                                                 (short)0, acc_out[et],
                                                                 false, false);
        }
        __syncthreads();
    }

    // Accumulate partial out across the JP column slices.
#pragma unroll
    for (int et = 0; et < 4; ++et) {
#pragma unroll
        for (int r = 0; r < 8; ++r) {
            const int i = i0 + r + 8 * h;
            const int e = 16 * et + m;
            atomicAdd(&outp[((size_t)bz * Sq + i) * Dq + e], acc_out[et][r]);
        }
    }
    // s_endpgm performs an implicit wait-idle, draining remaining async stores.
}

// ---- launcher --------------------------------------------------------------
extern "C" void kernel_launch(void* const* d_in, const int* in_sizes, int n_in,
                              void* d_out, int out_size, void* d_ws, size_t ws_size,
                              hipStream_t stream) {
    const float* qk   = (const float*)d_in[0];
    const float* v    = (const float*)d_in[1];
    const int*   msk  = (const int*)d_in[2];

    float* outp  = (float*)d_out;                          // [B,S,D]
    float* attnp = outp + (size_t)Bq * Sq * Dq;            // [B,S,S]

    char*     wsb  = (char*)d_ws;
    float*    rowM = (float*)wsb;                          // B*S floats
    float*    rowS = rowM + (size_t)Bq * Sq;               // B*S floats
    _Float16* kh   = (_Float16*)(wsb + 65536);             // B*S*D f16 (1 MB)
    _Float16* vt   = kh + (size_t)Bq * Sq * Dq;            // B*D*S f16 (1 MB)

    const int nOut = Bq * Sq * Dq;
    zero_out_kernel<<<(nOut + 255) / 256, 256, 0, stream>>>(outp, nOut);
    prep_kernel<<<Bq * Sq, 32, 0, stream>>>(qk, v, kh, vt);
    stats_kernel<<<(Bq * Sq / 16) / 4, 128, 0, stream>>>(qk, msk, kh, rowM, rowS);
    attn_out_kernel<<<(Bq * Sq / 16) * JP / 4, 128, 0, stream>>>(qk, msk, kh, vt,
                                                                 rowM, rowS, outp, attnp);
}
